// RNNModule_8546984919152
// MI455X (gfx1250) — matrix operand order repaired and checked by reading, not compiled, not run on hardware
//
#include <hip/hip_runtime.h>
#include <hip/hip_bf16.h>

typedef __attribute__((ext_vector_type(16))) __bf16 v16bf;
typedef __attribute__((ext_vector_type(8)))  float  v8f;

#define B_  64
#define T_  512
#define D_  1024
#define H_  1024
#define K_  2048           // H_ + D_ (concat dim)
#define NWG 64             // one 16-wide N stripe per workgroup
#define NTHREADS 256       // 8 waves: 4 M-tiles x 2 K-halves
#define LDS_STRIDE 2056    // 2048 + 8 halves pad (breaks 4096B bank stride)

// ---------------- fragment loaders ----------------
// A (16x32 bf16): lane m = lane%16; lanes 0-15 hold K {0..7,16..23}, 16-31 hold {8..15,24..31}
__device__ __forceinline__ v16bf load_a_bf16(const __bf16* row, int kbase, int kh8) {
  union { v16bf v; uint4 q[2]; } u;
  u.q[0] = *(const uint4*)(row + kbase + kh8);
  u.q[1] = *(const uint4*)(row + kbase + 16 + kh8);
  return u.v;
}

__device__ __forceinline__ v16bf load_a_f32(const float* row, int kbase, int kh8) {
  const float4* p = (const float4*)(row + kbase + kh8);
  const float4* q = (const float4*)(row + kbase + 16 + kh8);
  float4 f0 = p[0], f1 = p[1], f2 = q[0], f3 = q[1];
  v16bf v;
  v[0]=(__bf16)f0.x;  v[1]=(__bf16)f0.y;  v[2]=(__bf16)f0.z;  v[3]=(__bf16)f0.w;
  v[4]=(__bf16)f1.x;  v[5]=(__bf16)f1.y;  v[6]=(__bf16)f1.z;  v[7]=(__bf16)f1.w;
  v[8]=(__bf16)f2.x;  v[9]=(__bf16)f2.y;  v[10]=(__bf16)f2.z; v[11]=(__bf16)f2.w;
  v[12]=(__bf16)f3.x; v[13]=(__bf16)f3.y; v[14]=(__bf16)f3.z; v[15]=(__bf16)f3.w;
  return v;
}

// B (32x16 bf16): lane n = lane%16; lanes 0-15 hold K {0..7,16..23}, 16-31 hold {8..15,24..31}
__device__ __forceinline__ v16bf load_b_lds(const __bf16* wrow, int kbase, int kh16) {
  union { v16bf v; uint4 q[2]; } u;
  u.q[0] = *(const uint4*)(wrow + kbase + kh16 / 2);
  u.q[1] = *(const uint4*)(wrow + kbase + 16 + kh16 / 2);
  return u.v;
}

__device__ __forceinline__ v8f wmma_bf16(v16bf a, v16bf b, v8f c) {
  return __builtin_amdgcn_wmma_f32_16x16x32_bf16(false, a, false, b, (short)0, c, false, false);
}

// ---------------- grid-wide sense-reversing barrier ----------------
__device__ __forceinline__ void grid_barrier(unsigned* cnt, unsigned* gen) {
  __threadfence();
  __syncthreads();
  if (threadIdx.x == 0) {
    unsigned g = __hip_atomic_load(gen, __ATOMIC_RELAXED, __HIP_MEMORY_SCOPE_AGENT);
    unsigned v = __hip_atomic_fetch_add(cnt, 1u, __ATOMIC_ACQ_REL, __HIP_MEMORY_SCOPE_AGENT);
    if (v == NWG - 1) {
      __hip_atomic_store(cnt, 0u, __ATOMIC_RELAXED, __HIP_MEMORY_SCOPE_AGENT);
      __hip_atomic_fetch_add(gen, 1u, __ATOMIC_RELEASE, __HIP_MEMORY_SCOPE_AGENT);
    } else {
      while (__hip_atomic_load(gen, __ATOMIC_ACQUIRE, __HIP_MEMORY_SCOPE_AGENT) == g)
        __builtin_amdgcn_s_sleep(2);
    }
  }
  __syncthreads();
  __threadfence();
}

// ---------------- prep kernels ----------------
// W[K_][H_] f32 row-major -> Wt[H_][K_] bf16 (N-major, so B-fragments are contiguous)
__global__ void transpose_cvt_kernel(const float* __restrict__ W, __bf16* __restrict__ Wt) {
  __shared__ float tile[32][33];
  int kb = blockIdx.x * 32, nb = blockIdx.y * 32;
  int tx = threadIdx.x, ty = threadIdx.y;               // block (32,8)
  #pragma unroll
  for (int j = 0; j < 32; j += 8)
    tile[ty + j][tx] = W[(size_t)(kb + ty + j) * H_ + nb + tx];
  __syncthreads();
  #pragma unroll
  for (int j = 0; j < 32; j += 8)
    Wt[(size_t)(nb + ty + j) * K_ + kb + tx] = (__bf16)tile[tx][ty + j];
}

__global__ void init_kernel(__bf16* hstate, unsigned* cnt) {
  int i = blockIdx.x * blockDim.x + threadIdx.x;
  if (i < 2 * 2 * B_ * H_) hstate[i] = (__bf16)0.0f;    // h0[2 buf] + h1[2 buf]
  if (i == 0) *cnt = 0u;
}

// ---------------- persistent RNN kernel ----------------
__global__ void __launch_bounds__(NTHREADS, 1)
rnn_persistent(const float* __restrict__ x,
               const float* __restrict__ b0v,
               const float* __restrict__ b1v,
               const __bf16* __restrict__ Wt0,
               const __bf16* __restrict__ Wt1,
               __bf16* __restrict__ h0s,   // [2][B_][H_]
               __bf16* __restrict__ h1s,   // [2][B_][H_]
               float* __restrict__ out,
               unsigned* cnt, unsigned* gen) {
  __shared__ __align__(16) __bf16 ldsW[2][16][LDS_STRIDE];   // 128.5 KB weights
  __shared__ __align__(16) float  redbuf[4][32][8];          // 4 KB K-split partials

  const int tid  = threadIdx.x;
  const int wg   = blockIdx.x;
  const int n0   = wg * 16;                 // N stripe
  const int wave = tid >> 5;
  const int mt   = wave & 3;                // M tile 0..3
  const int grp  = wave >> 2;               // K-half: 0 = first 1024, 1 = second 1024
  const int lane = tid & 31;
  const int nl   = lane & 15;
  const int kh8  = (lane >> 4) * 8;
  const int kh16 = (lane >> 4) * 16;
  const int m0   = mt * 16;                 // M tile base
  const int mrow = m0 + nl;                 // A-row this lane loads
  const int mst  = m0 + (lane >> 4) * 8;    // C-row base this lane stores (+v)

  // Stage both layers' weight stripes into LDS once (reused all 512 steps)
  for (int layer = 0; layer < 2; ++layer) {
    const __bf16* Wt = layer ? Wt1 : Wt0;
    for (int i = tid; i < 16 * (K_ / 8); i += NTHREADS) {
      int r = i >> 8;                        // K_/8 == 256 chunks per row
      int c = (i & 255) * 8;
      *(uint4*)&ldsW[layer][r][c] = *(const uint4*)(Wt + (size_t)(n0 + r) * K_ + c);
    }
  }
  __syncthreads();

  const float bias0 = b0v[n0 + nl];
  const float bias1 = b1v[n0 + nl];
  const __bf16* wrow0 = &ldsW[0][nl][0];
  const __bf16* wrow1 = &ldsW[1][nl][0];
  const size_t buf = (size_t)B_ * H_;
  float* fin = out + (size_t)B_ * T_ * H_;  // final_state tail

  for (int t = 0; t < T_; ++t) {
    const int p = t & 1;                     // read buffer p, write 1-p
    const __bf16* h0rd = h0s + (size_t)p * buf + (size_t)mrow * H_;
    const __bf16* h1rd = h1s + (size_t)p * buf + (size_t)mrow * H_;
    __bf16* h0wr = h0s + (size_t)(1 - p) * buf;
    __bf16* h1wr = h1s + (size_t)(1 - p) * buf;
    const float* xrow = x + ((size_t)mrow * T_ + t) * D_;

    // ---------- layer 0: h0 = tanh([h0_prev, x_t] @ W0 + b0) ----------
    v8f acc;
    #pragma unroll
    for (int v = 0; v < 8; ++v) acc[v] = (grp == 0) ? bias0 : 0.0f;

    if (grp == 0) {                          // K half 0: h0_prev (bf16)
      #pragma unroll 4
      for (int k = 0; k < H_; k += 32)
        acc = wmma_bf16(load_a_bf16(h0rd, k, kh8), load_b_lds(wrow0, k, kh16), acc);
    } else {                                 // K half 1: x_t (f32 -> bf16 in-flight)
      #pragma unroll 4
      for (int k = 0; k < D_; k += 32)
        acc = wmma_bf16(load_a_f32(xrow, k, kh8), load_b_lds(wrow0, H_ + k, kh16), acc);
    }
    if (grp == 1) {
      union { v8f v; float4 f[2]; } u; u.v = acc;
      *(float4*)&redbuf[mt][lane][0] = u.f[0];
      *(float4*)&redbuf[mt][lane][4] = u.f[1];
    }
    __syncthreads();
    if (grp == 0) {
      union { v8f v; float4 f[2]; } r;
      r.f[0] = *(const float4*)&redbuf[mt][lane][0];
      r.f[1] = *(const float4*)&redbuf[mt][lane][4];
      float h0v[8];
      #pragma unroll
      for (int v = 0; v < 8; ++v) h0v[v] = tanhf(acc[v] + r.v[v]);
      #pragma unroll
      for (int v = 0; v < 8; ++v)
        h0wr[(size_t)(mst + v) * H_ + n0 + nl] = (__bf16)h0v[v];
      if (t == T_ - 1) {
        #pragma unroll
        for (int v = 0; v < 8; ++v)
          fin[(size_t)(mst + v) * (2 * H_) + n0 + nl] = h0v[v];
      }
    }
    grid_barrier(cnt, gen);

    // ---------- layer 1: h1 = tanh([h1_prev, h0_new] @ W1 + b1) ----------
    #pragma unroll
    for (int v = 0; v < 8; ++v) acc[v] = (grp == 0) ? bias1 : 0.0f;

    if (grp == 0) {                          // K half 0: h1_prev
      #pragma unroll 4
      for (int k = 0; k < H_; k += 32)
        acc = wmma_bf16(load_a_bf16(h1rd, k, kh8), load_b_lds(wrow1, k, kh16), acc);
    } else {                                 // K half 1: h0_new (written pre-barrier)
      // warm next timestep's x rows into cache (global_prefetch_b8 path)
      if (t + 1 < T_) {
        const float* xn = x + ((size_t)mrow * T_ + (t + 1)) * D_ + (size_t)(lane >> 4) * 512;
        #pragma unroll
        for (int j = 0; j < 4; ++j) __builtin_prefetch(xn + j * 128, 0, 1);
      }
      const __bf16* h0new = h0wr + (size_t)mrow * H_;
      #pragma unroll 4
      for (int k = 0; k < H_; k += 32)
        acc = wmma_bf16(load_a_bf16(h0new, k, kh8), load_b_lds(wrow1, H_ + k, kh16), acc);
    }
    if (grp == 1) {
      union { v8f v; float4 f[2]; } u; u.v = acc;
      *(float4*)&redbuf[mt][lane][0] = u.f[0];
      *(float4*)&redbuf[mt][lane][4] = u.f[1];
    }
    __syncthreads();
    if (grp == 0) {
      union { v8f v; float4 f[2]; } r;
      r.f[0] = *(const float4*)&redbuf[mt][lane][0];
      r.f[1] = *(const float4*)&redbuf[mt][lane][4];
      float h1v[8];
      #pragma unroll
      for (int v = 0; v < 8; ++v) h1v[v] = tanhf(acc[v] + r.v[v]);
      #pragma unroll
      for (int v = 0; v < 8; ++v) {
        h1wr[(size_t)(mst + v) * H_ + n0 + nl] = (__bf16)h1v[v];
        // outputs are write-once (128 MB): keep them out of L2's way
        __builtin_nontemporal_store(h1v[v], &out[((size_t)(mst + v) * T_ + t) * H_ + n0 + nl]);
      }
      if (t == T_ - 1) {
        #pragma unroll
        for (int v = 0; v < 8; ++v)
          fin[(size_t)(mst + v) * (2 * H_) + H_ + n0 + nl] = h1v[v];
      }
    }
    grid_barrier(cnt, gen);
  }
}

// ---------------- host ----------------
extern "C" void kernel_launch(void* const* d_in, const int* in_sizes, int n_in,
                              void* d_out, int out_size, void* d_ws, size_t ws_size,
                              hipStream_t stream) {
  const float* x  = (const float*)d_in[0];
  const float* W0 = (const float*)d_in[1];
  const float* b0 = (const float*)d_in[2];
  const float* W1 = (const float*)d_in[3];
  const float* b1 = (const float*)d_in[4];
  float* out = (float*)d_out;

  char* ws = (char*)d_ws;
  __bf16* Wt0 = (__bf16*)(ws);                                  // 4 MB
  __bf16* Wt1 = (__bf16*)(ws + (4u << 20));                     // 4 MB
  __bf16* h0s = (__bf16*)(ws + (8u << 20));                     // 256 KB (double-buffered)
  __bf16* h1s = (__bf16*)(ws + (8u << 20) + 262144u);           // 256 KB
  unsigned* cnt = (unsigned*)(ws + (8u << 20) + 524288u);
  unsigned* gen = cnt + 1;

  dim3 tb(32, 8);
  transpose_cvt_kernel<<<dim3(K_ / 32, H_ / 32), tb, 0, stream>>>(W0, Wt0);
  transpose_cvt_kernel<<<dim3(K_ / 32, H_ / 32), tb, 0, stream>>>(W1, Wt1);
  init_kernel<<<(2 * 2 * B_ * H_ + 255) / 256, 256, 0, stream>>>(h0s, cnt);
  rnn_persistent<<<NWG, NTHREADS, 0, stream>>>(x, b0, b1, Wt0, Wt1, h0s, h1s, out, cnt, gen);
}
